// GPTBlock_37091337568396
// MI455X (gfx1250) — matrix operand order repaired and checked by reading, not compiled, not run on hardware
//
#include <hip/hip_runtime.h>
#include <hip/hip_bf16.h>
#include <math.h>

// ---------------------------------------------------------------------------
// GPT block (pre-LN attn + MLP) for B=2, S=2048, E=2048, H=16, DH=128, DFF=8192
// bf16 WMMA (v_wmma_f32_16x16x32_bf16) everywhere, fp32 accumulate.
// GEMM wave tile 32x64 (2x4 WMMA accumulators) -> 1.5 b128 loads per WMMA.
// ---------------------------------------------------------------------------

typedef __bf16 bf16;
typedef __attribute__((ext_vector_type(16))) __bf16 v16bf;
typedef __attribute__((ext_vector_type(8)))  __bf16 v8bf;
typedef __attribute__((ext_vector_type(8)))  float  v8f;

#define B_    2
#define S_    2048
#define E_    2048
#define H_    16
#define DH_   128
#define DFF_  8192
#define TOKENS (B_*S_)
#define LN_EPS 1e-5f

union BF16x16 { v16bf v; v8bf h[2]; };

__device__ __forceinline__ v8bf ldg8(const bf16* p) { return *(const v8bf*)p; }

// --------------------------- f32 -> bf16 convert ---------------------------
__global__ __launch_bounds__(256) void cvt_f32_bf16(const float* __restrict__ in,
                                                    bf16* __restrict__ out, int n) {
    int i = blockIdx.x * 256 + threadIdx.x;
    if (i < n) out[i] = (bf16)in[i];
}

// --------------------------- LayerNorm -> bf16 -----------------------------
__global__ __launch_bounds__(256) void ln_to_bf16(const float* __restrict__ x,
                                                  const float* __restrict__ w,
                                                  const float* __restrict__ b,
                                                  bf16* __restrict__ out) {
    __shared__ float red[2][8];
    const int row = blockIdx.x;
    const float* xr = x + (size_t)row * E_;
    float vals[8];
    float s = 0.f, s2 = 0.f;
    for (int i = 0; i < 8; i++) {
        float v = xr[threadIdx.x + i * 256];
        vals[i] = v; s += v; s2 += v * v;
    }
    for (int m = 16; m >= 1; m >>= 1) {
        s  += __shfl_xor(s,  m, 32);
        s2 += __shfl_xor(s2, m, 32);
    }
    const int wave = threadIdx.x >> 5, lane = threadIdx.x & 31;
    if (lane == 0) { red[0][wave] = s; red[1][wave] = s2; }
    __syncthreads();
    s = 0.f; s2 = 0.f;
    for (int i = 0; i < 8; i++) { s += red[0][i]; s2 += red[1][i]; }
    const float mu   = s * (1.0f / E_);
    const float var  = s2 * (1.0f / E_) - mu * mu;
    const float rinv = rsqrtf(var + LN_EPS);
    for (int i = 0; i < 8; i++) {
        int c = threadIdx.x + i * 256;
        out[(size_t)row * E_ + c] = (bf16)((vals[i] - mu) * rinv * w[c] + b[c]);
    }
}

// ------------------------- V transpose (for attn) --------------------------
// qkv[bs, 2E + h*DH + d] -> vT[((b*H + h)*DH + d)*S + s]
__global__ __launch_bounds__(256) void transpose_v(const bf16* __restrict__ qkv,
                                                   bf16* __restrict__ vT) {
    size_t i = (size_t)blockIdx.x * 256 + threadIdx.x;   // over B*S*E
    int    c  = (int)(i % E_);
    size_t bs = i / E_;
    int b = (int)(bs / S_), s = (int)(bs % S_);
    int hh = c / DH_, d = c % DH_;
    vT[((size_t)(b * H_ + hh) * DH_ + d) * S_ + s] = qkv[bs * (3 * E_) + 2 * E_ + c];
}

// ------------------------------ WMMA GEMM ----------------------------------
// D[M,N] = A[M,K](bf16,row) * W[N,K](bf16,row)^T + bias
// Block: 8 waves (2 row-groups x 4 col-groups), block tile 64 x 256.
// Wave tile 32x64: 2x4 accumulators; 12 b128 loads per 8 WMMAs per K=32 step.
// MODE 1: store bf16      MODE 2: +resid(f32), store f32      MODE 3: GELU, store bf16
template <int MODE>
__global__ __launch_bounds__(256)
void gemm_bf16(const bf16* __restrict__ A, const bf16* __restrict__ W,
               const float* __restrict__ bias, const float* __restrict__ resid,
               float* __restrict__ outF, bf16* __restrict__ outB,
               int M, int N, int K) {
    const int lane = threadIdx.x & 31;
    const int wave = threadIdx.x >> 5;
    const int lm = lane & 15, h = lane >> 4;
    const int wm = wave & 1, wn = wave >> 1;            // 2 x 4 wave grid
    const int row0 = blockIdx.y * 64  + wm * 32;
    const int col0 = blockIdx.x * 256 + wn * 64;

    v8f c[2][4] = {};
    const bf16* aBase[2];
    const bf16* bBase[4];
    for (int i = 0; i < 2; i++)
        aBase[i] = A + (size_t)(row0 + i * 16 + lm) * K + h * 8;    // A-frag layout
    for (int j = 0; j < 4; j++)
        bBase[j] = W + (size_t)(col0 + j * 16 + lm) * K + h * 8;    // B-frag layout

    for (int kt = 0; kt < K; kt += 32) {
        // pull next K-tile toward the WGP cache (global_prefetch_b8)
        __builtin_prefetch(aBase[0] + kt + 32, 0, 3);
        __builtin_prefetch(bBase[0] + kt + 32, 0, 3);

        BF16x16 a[2], b[4];
        for (int i = 0; i < 2; i++) {
            a[i].h[0] = ldg8(aBase[i] + kt);
            a[i].h[1] = ldg8(aBase[i] + kt + 16);
        }
        for (int j = 0; j < 4; j++) {
            b[j].h[0] = ldg8(bBase[j] + kt);
            b[j].h[1] = ldg8(bBase[j] + kt + 16);
        }
        for (int i = 0; i < 2; i++)
            for (int j = 0; j < 4; j++)
                c[i][j] = __builtin_amdgcn_wmma_f32_16x16x32_bf16(
                    false, a[i].v, false, b[j].v, (short)0, c[i][j], false, false);
    }

    for (int i = 0; i < 2; i++) {
        for (int j = 0; j < 4; j++) {
            const int col = col0 + j * 16 + lm;
            const float bv = bias[col];
            for (int r = 0; r < 8; r++) {
                const int row = row0 + i * 16 + h * 8 + r;
                float v = c[i][j][r] + bv;
                if (MODE == 2) v += resid[(size_t)row * N + col];
                if (MODE == 3) v = 0.5f * v * (1.0f + erff(v * 0.70710678118654752f));
                if (MODE == 2) outF[(size_t)row * N + col] = v;
                else           outB[(size_t)row * N + col] = (bf16)v;
            }
        }
    }
}

// --------------------------- Flash attention -------------------------------
// 1 wave handles 16 query rows for one (b,h). 4 waves / block.
__global__ __launch_bounds__(128)
void attn_fwd(const bf16* __restrict__ qkv, const bf16* __restrict__ vT,
              const unsigned char* __restrict__ kpm, bf16* __restrict__ ctxOut) {
    __shared__ __align__(16) bf16 pT[4][16][32];
    const int lane = threadIdx.x & 31, wave = threadIdx.x >> 5;
    const int lm = lane & 15, h = lane >> 4;
    const int bh = blockIdx.y;
    const int b = bh / H_, hh = bh % H_;
    const int q0 = blockIdx.x * 64 + wave * 16;
    const float scale = 0.08838834764831845f;   // 1/sqrt(128)

    // persistent Q A-fragments (16 rows x 128 = 4 frags of K=32)
    BF16x16 qf[4];
    const bf16* qptr = qkv + (size_t)(b * S_ + q0 + lm) * (3 * E_) + hh * DH_ + h * 8;
    for (int f = 0; f < 4; f++) {
        qf[f].h[0] = ldg8(qptr + f * 32);
        qf[f].h[1] = ldg8(qptr + f * 32 + 16);
    }

    v8f acc[8] = {};
    float mrow[8], lrow[8];
    for (int r = 0; r < 8; r++) { mrow[r] = -1e30f; lrow[r] = 0.f; }

    const bf16* kbase = qkv + (size_t)(b * S_) * (3 * E_) + E_ + hh * DH_;
    const bf16* vbase = vT + (size_t)(b * H_ + hh) * DH_ * S_;

    for (int kb = 0; kb <= q0 + 15; kb += 32) {
        // scores: two 16x16 tiles covering 32 keys
        v8f sc[2] = {};
        for (int t = 0; t < 2; t++) {
            const bf16* kp = kbase + (size_t)(kb + t * 16 + lm) * (3 * E_) + h * 8;
            for (int f = 0; f < 4; f++) {
                BF16x16 bfr;
                bfr.h[0] = ldg8(kp + f * 32);
                bfr.h[1] = ldg8(kp + f * 32 + 16);
                sc[t] = __builtin_amdgcn_wmma_f32_16x16x32_bf16(
                    false, qf[f].v, false, bfr.v, (short)0, sc[t], false, false);
            }
        }
        // scale + causal + key-padding mask (C layout: lane owns key, rows r+8h)
        float p[2][8];
        for (int t = 0; t < 2; t++) {
            const int key = kb + t * 16 + lm;
            const bool kok = kpm[b * S_ + key] != 0;
            for (int r = 0; r < 8; r++) {
                const int qr = q0 + h * 8 + r;
                float v = sc[t][r] * scale;
                if (key > qr || !kok) v = -1e30f;
                p[t][r] = v;
            }
        }
        // online softmax per row (reduce across 16 lanes of the half-wave)
        for (int r = 0; r < 8; r++) {
            float mx = fmaxf(p[0][r], p[1][r]);
            for (int m = 1; m < 16; m <<= 1) mx = fmaxf(mx, __shfl_xor(mx, m, 32));
            const float mnew  = fmaxf(mrow[r], mx);
            const float alpha = __expf(mrow[r] - mnew);
            const float e0 = __expf(p[0][r] - mnew);
            const float e1 = __expf(p[1][r] - mnew);
            p[0][r] = e0; p[1][r] = e1;
            float rs = e0 + e1;
            for (int m = 1; m < 16; m <<= 1) rs += __shfl_xor(rs, m, 32);
            lrow[r] = lrow[r] * alpha + rs;
            mrow[r] = mnew;
            for (int nt = 0; nt < 8; nt++) acc[nt][r] *= alpha;
        }
        // C-layout -> A-layout transpose of P through per-wave LDS tile
        for (int t = 0; t < 2; t++)
            for (int r = 0; r < 8; r++)
                pT[wave][h * 8 + r][t * 16 + lm] = (bf16)p[t][r];
        asm volatile("s_wait_dscnt 0" ::: "memory");
        BF16x16 pf;
        pf.h[0] = *(const v8bf*)&pT[wave][lm][h * 8];
        pf.h[1] = *(const v8bf*)&pT[wave][lm][16 + h * 8];
        // ctx += P(16x32) * V(32x128): 8 WMMAs over DH tiles; Vt is [dh][s]
        for (int nt = 0; nt < 8; nt++) {
            const bf16* vp = vbase + (size_t)(nt * 16 + lm) * S_ + kb + h * 8;
            BF16x16 vf;
            vf.h[0] = ldg8(vp);
            vf.h[1] = ldg8(vp + 16);
            acc[nt] = __builtin_amdgcn_wmma_f32_16x16x32_bf16(
                false, pf.v, false, vf.v, (short)0, acc[nt], false, false);
        }
    }
    // normalize and write ctx (bf16, [B*S, E], col = h*DH + d)
    for (int nt = 0; nt < 8; nt++) {
        for (int r = 0; r < 8; r++) {
            const int row = q0 + h * 8 + r;
            const int col = hh * DH_ + nt * 16 + lm;
            ctxOut[(size_t)(b * S_ + row) * E_ + col] = (bf16)(acc[nt][r] / lrow[r]);
        }
    }
}

// ------------------------------- launcher ----------------------------------
extern "C" void kernel_launch(void* const* d_in, const int* in_sizes, int n_in,
                              void* d_out, int out_size, void* d_ws, size_t ws_size,
                              hipStream_t stream) {
    const float* x        = (const float*)d_in[0];
    const unsigned char* kpm = (const unsigned char*)d_in[1];
    // d_in[2] = causal attn_mask (applied analytically)
    const float* ln1_w    = (const float*)d_in[3];
    const float* ln1_b    = (const float*)d_in[4];
    const float* in_proj_w= (const float*)d_in[5];
    const float* in_proj_b= (const float*)d_in[6];
    const float* out_w    = (const float*)d_in[7];
    const float* out_b    = (const float*)d_in[8];
    const float* ln2_w    = (const float*)d_in[9];
    const float* ln2_b    = (const float*)d_in[10];
    const float* up_w     = (const float*)d_in[11];
    const float* up_b     = (const float*)d_in[12];
    const float* down_w   = (const float*)d_in[13];
    const float* down_b   = (const float*)d_in[14];

    char* ws = (char*)d_ws;
    size_t off = 0;
    bf16* wQKV = (bf16*)(ws + off); off += (size_t)3 * E_ * E_ * 2;   // 24 MiB
    bf16* wO   = (bf16*)(ws + off); off += (size_t)E_ * E_ * 2;       //  8 MiB
    bf16* wU   = (bf16*)(ws + off); off += (size_t)DFF_ * E_ * 2;     // 32 MiB
    bf16* wD   = (bf16*)(ws + off); off += (size_t)E_ * DFF_ * 2;     // 32 MiB
    bf16* aB   = (bf16*)(ws + off); off += (size_t)TOKENS * E_ * 2;   // 16 MiB
    bf16* qkvB = (bf16*)(ws + off); off += (size_t)TOKENS * 3 * E_ * 2; // 48 MiB
    bf16* vTB  = (bf16*)(ws + off); off += (size_t)TOKENS * E_ * 2;   // 16 MiB
    bf16* ctxB = (bf16*)(ws + off); off += (size_t)TOKENS * E_ * 2;   // 16 MiB
    float* x1  = (float*)(ws + off); off += (size_t)TOKENS * E_ * 4;  // 32 MiB
    bf16* mB   = (bf16*)(ws + off); off += (size_t)TOKENS * E_ * 2;   // 16 MiB
    bf16* hB   = (bf16*)(ws + off); off += (size_t)TOKENS * DFF_ * 2; // 64 MiB

    // weights -> bf16 (stay resident in 192MB L2 across the GEMMs)
    {
        int n;
        n = 3 * E_ * E_;  cvt_f32_bf16<<<(n + 255) / 256, 256, 0, stream>>>(in_proj_w, wQKV, n);
        n = E_ * E_;      cvt_f32_bf16<<<(n + 255) / 256, 256, 0, stream>>>(out_w,     wO,   n);
        n = DFF_ * E_;    cvt_f32_bf16<<<(n + 255) / 256, 256, 0, stream>>>(up_w,      wU,   n);
        n = E_ * DFF_;    cvt_f32_bf16<<<(n + 255) / 256, 256, 0, stream>>>(down_w,    wD,   n);
    }

    // 1) a = LN1(x) -> bf16
    ln_to_bf16<<<TOKENS, 256, 0, stream>>>(x, ln1_w, ln1_b, aB);
    // 2) qkv = a @ in_proj_w^T + b  -> bf16 [TOKENS, 3E]
    gemm_bf16<1><<<dim3((3 * E_) / 256, TOKENS / 64), 256, 0, stream>>>(
        aB, wQKV, in_proj_b, nullptr, nullptr, qkvB, TOKENS, 3 * E_, E_);
    // 3) V -> Vt [B,H,DH,S]
    transpose_v<<<(unsigned)((size_t)TOKENS * E_ / 256), 256, 0, stream>>>(qkvB, vTB);
    // 4) causal flash attention -> ctx bf16 [TOKENS, E]
    attn_fwd<<<dim3(S_ / 64, B_ * H_), 128, 0, stream>>>(qkvB, vTB, kpm, ctxB);
    // 5) x1 = x + ctx @ out_w^T + out_b   (f32)
    gemm_bf16<2><<<dim3(E_ / 256, TOKENS / 64), 256, 0, stream>>>(
        ctxB, wO, out_b, x, x1, nullptr, TOKENS, E_, E_);
    // 6) m = LN2(x1) -> bf16
    ln_to_bf16<<<TOKENS, 256, 0, stream>>>(x1, ln2_w, ln2_b, mB);
    // 7) hB = GELU(m @ up_w^T + up_b) -> bf16 [TOKENS, DFF]
    gemm_bf16<3><<<dim3(DFF_ / 256, TOKENS / 64), 256, 0, stream>>>(
        mB, wU, up_b, nullptr, nullptr, hB, TOKENS, DFF_, E_);
    // 8) out = x1 + hB @ down_w^T + down_b  (f32, final output)
    gemm_bf16<2><<<dim3(E_ / 256, TOKENS / 64), 256, 0, stream>>>(
        hB, wD, down_b, x1, (float*)d_out, nullptr, TOKENS, E_, DFF_);
}
